// TreeLSTM_9208409883266
// MI455X (gfx1250) — compile-verified, matrix operand order checked by reading, not run on hardware
//
#include <hip/hip_runtime.h>
#include <cstdint>

// ---------------------------------------------------------------------------
// TreeLSTM forward on MI455X (gfx1250, wave32, WMMA bf16 16x16x32)
// ---------------------------------------------------------------------------
#define BATCH  128
#define NTREEN 1023
#define NLEAF  512
#define HS     256
#define H3     768
#define VOCAB  32000

typedef __bf16 v16bf __attribute__((ext_vector_type(16)));
typedef float  v8f   __attribute__((ext_vector_type(8)));

union FragBF { float4 f4[2]; v16bf v; };
union U16x8  { float4 f4; unsigned short s[8]; };

__device__ __forceinline__ unsigned short f2bf(float x) {
  unsigned int u = __float_as_uint(x);
  u += 0x7FFFu + ((u >> 16) & 1u);               // round-to-nearest-even
  return (unsigned short)(u >> 16);
}
__device__ __forceinline__ float bf2f(unsigned short b) {
  return __uint_as_float(((unsigned int)b) << 16);
}
__device__ __forceinline__ float sigm(float x) { return 1.0f / (1.0f + __expf(-x)); }

__device__ __forceinline__ v8f vzero8() {
  v8f v;
#pragma unroll
  for (int i = 0; i < 8; ++i) v[i] = 0.0f;
  return v;
}

// Load one 16x32 bf16 fragment (A layout; B uses identical packing with N as
// the "row") from an LDS tile with row stride 32 elements.
// lane 0-15: row = base+lane, K in {kb..kb+7} U {16+kb..16+kb+7}, kb = (lane>>4)*8
__device__ __forceinline__ v16bf ld_frag(const unsigned short* tile, int row16, int lane) {
  const unsigned short* p = tile + (row16 + (lane & 15)) * 32;
  const int kb = (lane >> 4) << 3;
  FragBF f;
  f.f4[0] = *(const float4*)(p + kb);
  f.f4[1] = *(const float4*)(p + 16 + kb);
  return f.v;
}

__device__ __forceinline__ v8f wmma_bf16(v16bf a, v16bf b, v8f c) {
  return __builtin_amdgcn_wmma_f32_16x16x32_bf16(false, a, false, b, (short)0, c, false, false);
}

// ---------------------------------------------------------------------------
// f32 -> bf16 conversion (weights + embedding table)
// ---------------------------------------------------------------------------
__global__ void cvt_bf16_kernel(const float* __restrict__ src,
                                unsigned short* __restrict__ dst, int n) {
  int i = blockIdx.x * blockDim.x + threadIdx.x;
  if (i < n) dst[i] = f2bf(src[i]);
}

// ---------------------------------------------------------------------------
// Leaf kernel: iou = gather(emb)[.] @ W_iou^T + b_iou ; LSTM cell with c_sum=0
// Block: 128 rows x 64 h-cols (x3 gates). 8 waves, wave tile 32x32 per gate.
// grid.x = 1024 (512 blocks per tree), grid.y = 4
// ---------------------------------------------------------------------------
__global__ __launch_bounds__(256) void leaf_kernel(
    const int* __restrict__ types1, const int* __restrict__ types2,
    const unsigned short* __restrict__ embb,
    const unsigned short* __restrict__ Wb,      // [768][256] bf16
    const float* __restrict__ biou,             // [768]
    unsigned short* __restrict__ hout,          // [2*NTREE*B][256] bf16
    float* __restrict__ cout)                   // [2*NTREE*B][256] f32
{
  __shared__ unsigned short As[128 * 32];
  __shared__ unsigned short Bs[3 * 64 * 32];

  const int tid  = threadIdx.x;
  const int lane = tid & 31;
  const int wid  = tid >> 5;
  const int wm   = wid & 3;            // 4 row slots of 32
  const int wn   = wid >> 2;           // 2 col slots of 32
  const int tree = blockIdx.x >> 9;    // 512 row-blocks per tree
  const int mt0  = (blockIdx.x & 511) * 128;
  const int nb0  = blockIdx.y * 64;

  const int* types = tree ? types2 : types1;

  // Per-thread A staging address (row is K-invariant)
  const int arow  = tid >> 1;
  const int ahalf = tid & 1;
  const int mt    = mt0 + arow;
  const int leaf  = mt >> 7;
  const int bidx  = mt & 127;
  const int typ   = types[bidx * NTREEN + (NLEAF - 1 + leaf)];
  const unsigned short* asrc0 = embb + (size_t)typ * 256 + ahalf * 16;

  v8f acc[3][2][2];
#pragma unroll
  for (int g = 0; g < 3; ++g)
#pragma unroll
    for (int a = 0; a < 2; ++a)
#pragma unroll
      for (int b = 0; b < 2; ++b) acc[g][a][b] = vzero8();

  for (int kt = 0; kt < 8; ++kt) {
    const int k0 = kt * 32;
    // stage A (gathered bf16 embedding rows): 32 bytes / thread
    {
      const float4* s = (const float4*)(asrc0 + k0);
      float4* d = (float4*)(As + arow * 32 + ahalf * 16);
      d[0] = s[0];
      d[1] = s[1];
      if (kt < 7) __builtin_prefetch(asrc0 + k0 + 32, 0, 1);
    }
    // stage B: 3 gates x 64 n-rows x 32 k
    for (int i = tid; i < 384; i += 256) {
      const int gate = i >> 7;
      const int rem  = i & 127;
      const int nrow = rem >> 1;
      const int half = rem & 1;
      const int n    = gate * 256 + nb0 + nrow;
      const float4* s = (const float4*)(Wb + (size_t)n * 256 + k0 + half * 16);
      float4* d = (float4*)(Bs + (gate * 64 + nrow) * 32 + half * 16);
      d[0] = s[0];
      d[1] = s[1];
    }
    __syncthreads();

    v16bf af0 = ld_frag(As, wm * 32, lane);
    v16bf af1 = ld_frag(As, wm * 32 + 16, lane);
#pragma unroll
    for (int g = 0; g < 3; ++g) {
      v16bf b0 = ld_frag(Bs + g * 64 * 32, wn * 32, lane);
      v16bf b1 = ld_frag(Bs + g * 64 * 32, wn * 32 + 16, lane);
      acc[g][0][0] = wmma_bf16(af0, b0, acc[g][0][0]);
      acc[g][1][0] = wmma_bf16(af1, b0, acc[g][1][0]);
      acc[g][0][1] = wmma_bf16(af0, b1, acc[g][0][1]);
      acc[g][1][1] = wmma_bf16(af1, b1, acc[g][1][1]);
    }
    __syncthreads();
  }

  // Epilogue: cell, store h (bf16) and c (f32)
#pragma unroll
  for (int im = 0; im < 2; ++im)
#pragma unroll
    for (int jn = 0; jn < 2; ++jn)
#pragma unroll
      for (int r = 0; r < 8; ++r) {
        const int mloc = wm * 32 + im * 16 + r + ((lane >> 4) << 3);
        const int n    = nb0 + wn * 32 + jn * 16 + (lane & 15);
        const int mte  = mt0 + mloc;
        const int lf   = mte >> 7;
        const int b    = mte & 127;
        const size_t hrow = ((size_t)tree * NTREEN + (NLEAF - 1 + lf)) * BATCH + b;
        const float iv = acc[0][im][jn][r] + biou[n];
        const float ov = acc[1][im][jn][r] + biou[256 + n];
        const float uv = acc[2][im][jn][r] + biou[512 + n];
        const float cn = sigm(iv) * tanhf(uv);
        const float hn = sigm(ov) * tanhf(cn);
        hout[hrow * 256 + n] = f2bf(hn);
        cout[hrow * 256 + n] = cn;
      }
}

// ---------------------------------------------------------------------------
// Forget-gate kernel (per level): fc = sigmoid(h_child @ U_f^T + b_f) * c_child
// Rows enumerate (tree, childLocal, b) over the child level. BM=128, BN=128.
// grid.x = 2*nchild, grid.y = 2
// ---------------------------------------------------------------------------
__global__ __launch_bounds__(256) void fgate_kernel(
    const unsigned short* __restrict__ hbuf, const float* __restrict__ cbuf,
    const unsigned short* __restrict__ Ufb,   // [256][256] bf16
    const float* __restrict__ Ufbias,         // [256]
    float* __restrict__ fc,                   // [2*nchild*B][256]
    int nchild, int node0c)
{
  __shared__ unsigned short As[128 * 32];
  __shared__ unsigned short Bs[128 * 32];

  const int tid  = threadIdx.x;
  const int lane = tid & 31;
  const int wid  = tid >> 5;
  const int wm   = wid & 1;            // 2 row slots of 64
  const int wn   = wid >> 1;           // 4 col slots of 32
  const int tree = blockIdx.x / nchild;
  const int mt0  = (blockIdx.x % nchild) * 128;
  const int nb0  = blockIdx.y * 128;

  const int arow  = tid >> 1;
  const int ahalf = tid & 1;
  const int mt    = mt0 + arow;
  const size_t hrow_stage = ((size_t)tree * NTREEN + node0c + (mt >> 7)) * BATCH + (mt & 127);

  v8f acc[4][2];
#pragma unroll
  for (int a = 0; a < 4; ++a)
#pragma unroll
    for (int b = 0; b < 2; ++b) acc[a][b] = vzero8();

  for (int kt = 0; kt < 8; ++kt) {
    const int k0 = kt * 32;
    {
      const float4* s = (const float4*)(hbuf + hrow_stage * 256 + k0 + ahalf * 16);
      float4* d = (float4*)(As + arow * 32 + ahalf * 16);
      d[0] = s[0];
      d[1] = s[1];
    }
    {
      const int nrow = tid >> 1;
      const int half = tid & 1;
      const float4* s = (const float4*)(Ufb + (size_t)(nb0 + nrow) * 256 + k0 + half * 16);
      float4* d = (float4*)(Bs + nrow * 32 + half * 16);
      d[0] = s[0];
      d[1] = s[1];
    }
    __syncthreads();

    v16bf bf0 = ld_frag(Bs, wn * 32, lane);
    v16bf bf1 = ld_frag(Bs, wn * 32 + 16, lane);
#pragma unroll
    for (int im = 0; im < 4; ++im) {
      v16bf af = ld_frag(As, wm * 64 + im * 16, lane);
      acc[im][0] = wmma_bf16(af, bf0, acc[im][0]);
      acc[im][1] = wmma_bf16(af, bf1, acc[im][1]);
    }
    __syncthreads();
  }

#pragma unroll
  for (int im = 0; im < 4; ++im)
#pragma unroll
    for (int jn = 0; jn < 2; ++jn)
#pragma unroll
      for (int r = 0; r < 8; ++r) {
        const int mloc = wm * 64 + im * 16 + r + ((lane >> 4) << 3);
        const int n    = nb0 + wn * 32 + jn * 16 + (lane & 15);
        const int mte  = mt0 + mloc;
        const size_t hrow = ((size_t)tree * NTREEN + node0c + (mte >> 7)) * BATCH + (mte & 127);
        const float f = sigm(acc[im][jn][r] + Ufbias[n]);
        const size_t fcrow = (size_t)tree * nchild * BATCH + mte;
        fc[fcrow * 256 + n] = f * cbuf[hrow * 256 + n];
      }
}

// ---------------------------------------------------------------------------
// Internal iou kernel (per level):
//   h_tild = h[c0]+h[c1] (bf16 sum during staging);  iou = h_tild @ U_iou^T + b
//   c = sig(i)*tanh(u) + fc[c0]+fc[c1];  h = sig(o)*tanh(c)
// grid.x = 2*nnodes, grid.y = 4
// ---------------------------------------------------------------------------
__global__ __launch_bounds__(256) void iou_kernel(
    const unsigned short* __restrict__ hbuf, const float* __restrict__ fc,
    const unsigned short* __restrict__ Ub,    // [768][256] bf16
    const float* __restrict__ biou,
    unsigned short* __restrict__ hout, float* __restrict__ cout,
    int nnodes, int node0, int node0c)
{
  __shared__ unsigned short As[128 * 32];
  __shared__ unsigned short Bs[3 * 64 * 32];

  const int tid  = threadIdx.x;
  const int lane = tid & 31;
  const int wid  = tid >> 5;
  const int wm   = wid & 3;
  const int wn   = wid >> 2;
  const int tree = blockIdx.x / nnodes;
  const int mt0  = (blockIdx.x % nnodes) * 128;
  const int nb0  = blockIdx.y * 64;
  const int nchild = 2 * nnodes;

  const int arow  = tid >> 1;
  const int ahalf = tid & 1;
  const int mt    = mt0 + arow;
  const int jnode = mt >> 7;
  const int bidx  = mt & 127;
  const size_t r0 = ((size_t)tree * NTREEN + node0c + 2 * jnode) * BATCH + bidx;
  const size_t r1 = r0 + BATCH;   // sibling node is +1 -> +BATCH rows

  v8f acc[3][2][2];
#pragma unroll
  for (int g = 0; g < 3; ++g)
#pragma unroll
    for (int a = 0; a < 2; ++a)
#pragma unroll
      for (int b = 0; b < 2; ++b) acc[g][a][b] = vzero8();

  for (int kt = 0; kt < 8; ++kt) {
    const int k0 = kt * 32;
    // stage A = h[c0] + h[c1] (16 bf16 per thread)
    {
      unsigned short* d = As + arow * 32 + ahalf * 16;
#pragma unroll
      for (int v = 0; v < 2; ++v) {
        U16x8 x0, x1, o;
        x0.f4 = *(const float4*)(hbuf + r0 * 256 + k0 + ahalf * 16 + v * 8);
        x1.f4 = *(const float4*)(hbuf + r1 * 256 + k0 + ahalf * 16 + v * 8);
#pragma unroll
        for (int t = 0; t < 8; ++t) o.s[t] = f2bf(bf2f(x0.s[t]) + bf2f(x1.s[t]));
        *(float4*)(d + v * 8) = o.f4;
      }
    }
    // stage B: 3 gates x 64 n-rows
    for (int i = tid; i < 384; i += 256) {
      const int gate = i >> 7;
      const int rem  = i & 127;
      const int nrow = rem >> 1;
      const int half = rem & 1;
      const int n    = gate * 256 + nb0 + nrow;
      const float4* s = (const float4*)(Ub + (size_t)n * 256 + k0 + half * 16);
      float4* d = (float4*)(Bs + (gate * 64 + nrow) * 32 + half * 16);
      d[0] = s[0];
      d[1] = s[1];
    }
    __syncthreads();

    v16bf af0 = ld_frag(As, wm * 32, lane);
    v16bf af1 = ld_frag(As, wm * 32 + 16, lane);
#pragma unroll
    for (int g = 0; g < 3; ++g) {
      v16bf b0 = ld_frag(Bs + g * 64 * 32, wn * 32, lane);
      v16bf b1 = ld_frag(Bs + g * 64 * 32, wn * 32 + 16, lane);
      acc[g][0][0] = wmma_bf16(af0, b0, acc[g][0][0]);
      acc[g][1][0] = wmma_bf16(af1, b0, acc[g][1][0]);
      acc[g][0][1] = wmma_bf16(af0, b1, acc[g][0][1]);
      acc[g][1][1] = wmma_bf16(af1, b1, acc[g][1][1]);
    }
    __syncthreads();
  }

#pragma unroll
  for (int im = 0; im < 2; ++im)
#pragma unroll
    for (int jn = 0; jn < 2; ++jn)
#pragma unroll
      for (int r = 0; r < 8; ++r) {
        const int mloc = wm * 32 + im * 16 + r + ((lane >> 4) << 3);
        const int n    = nb0 + wn * 32 + jn * 16 + (lane & 15);
        const int mte  = mt0 + mloc;
        const int j    = mte >> 7;
        const int b    = mte & 127;
        const size_t prow  = ((size_t)tree * NTREEN + node0 + j) * BATCH + b;
        const size_t fcr0  = ((size_t)tree * nchild + 2 * j) * BATCH + b;
        const float csum = fc[fcr0 * 256 + n] + fc[(fcr0 + BATCH) * 256 + n];
        const float iv = acc[0][im][jn][r] + biou[n];
        const float ov = acc[1][im][jn][r] + biou[256 + n];
        const float uv = acc[2][im][jn][r] + biou[512 + n];
        const float cn = sigm(iv) * tanhf(uv) + csum;
        const float hn = sigm(ov) * tanhf(cn);
        hout[prow * 256 + n] = f2bf(hn);
        cout[prow * 256 + n] = cn;
      }
}

// ---------------------------------------------------------------------------
// rep[tree,b,:] = mean over 1023 nodes of h   (grid = 256 blocks, 256 thr)
// ---------------------------------------------------------------------------
__global__ __launch_bounds__(256) void reduce_kernel(
    const unsigned short* __restrict__ h, float* __restrict__ rep)
{
  const int tb = blockIdx.x;           // tree*128 + b
  const int tree = tb >> 7;
  const int b = tb & 127;
  const int n = threadIdx.x;
  float s = 0.0f;
  for (int node = 0; node < NTREEN; ++node)
    s += bf2f(h[(((size_t)tree * NTREEN + node) * BATCH + b) * 256 + n]);
  rep[(size_t)tb * 256 + n] = s * (1.0f / (float)NTREEN);
}

// ---------------------------------------------------------------------------
// out[b,c] = |rep1-rep2| @ cls_w^T + cls_b   (1 block, 128 thr)
// ---------------------------------------------------------------------------
__global__ void cls_kernel(const float* __restrict__ rep,
                           const float* __restrict__ w,
                           const float* __restrict__ bias,
                           float* __restrict__ out)
{
  const int b = threadIdx.x;
  float a0 = 0.0f, a1 = 0.0f;
  for (int n = 0; n < 256; ++n) {
    const float d = fabsf(rep[b * 256 + n] - rep[(128 + b) * 256 + n]);
    a0 += d * w[n];
    a1 += d * w[256 + n];
  }
  out[b * 2 + 0] = a0 + bias[0];
  out[b * 2 + 1] = a1 + bias[1];
}

// ---------------------------------------------------------------------------
extern "C" void kernel_launch(void* const* d_in, const int* in_sizes, int n_in,
                              void* d_out, int out_size, void* d_ws, size_t ws_size,
                              hipStream_t stream) {
  const int*   types1 = (const int*)d_in[0];
  const int*   types2 = (const int*)d_in[1];
  const float* emb    = (const float*)d_in[2];
  const float* W_iou  = (const float*)d_in[3];
  const float* U_iou  = (const float*)d_in[4];
  const float* b_iou  = (const float*)d_in[5];
  const float* U_f_w  = (const float*)d_in[6];
  const float* U_f_b  = (const float*)d_in[7];
  const float* cls_w  = (const float*)d_in[8];
  const float* cls_b  = (const float*)d_in[9];
  float* out = (float*)d_out;

  // Workspace layout (bytes). Requires ws_size >= ~530 MB.
  char* ws = (char*)d_ws;
  unsigned short* embb = (unsigned short*)(ws);                    // 16,384,000
  unsigned short* Wb   = (unsigned short*)(ws + 16384000);         //    393,216
  unsigned short* Ub   = (unsigned short*)(ws + 16777216);         //    393,216
  unsigned short* Ufb  = (unsigned short*)(ws + 17170432);         //    131,072
  unsigned short* hbuf = (unsigned short*)(ws + 17301504);         // 134,086,656
  float*          cbuf = (float*)(ws + 151388160);                 // 268,173,312
  float*          fc   = (float*)(ws + 419561472);                 // 134,217,728
  float*          rep  = (float*)(ws + 553779200);                 //    262,144

  // 1) convert weights + embedding table to bf16
  cvt_bf16_kernel<<<(VOCAB * 256 + 255) / 256, 256, 0, stream>>>(emb, embb, VOCAB * 256);
  cvt_bf16_kernel<<<(H3 * HS + 255) / 256, 256, 0, stream>>>(W_iou, Wb, H3 * HS);
  cvt_bf16_kernel<<<(H3 * HS + 255) / 256, 256, 0, stream>>>(U_iou, Ub, H3 * HS);
  cvt_bf16_kernel<<<(HS * HS + 255) / 256, 256, 0, stream>>>(U_f_w, Ufb, HS * HS);

  // 2) leaf level (both trees): 1024 x 4 blocks
  leaf_kernel<<<dim3(1024, 4), 256, 0, stream>>>(types1, types2, embb, Wb, b_iou,
                                                 hbuf, cbuf);

  // 3) internal levels, leaves -> root
  for (int lvl = 8; lvl >= 0; --lvl) {
    const int nnodes = 1 << lvl;
    const int node0  = nnodes - 1;
    const int nchild = 2 * nnodes;
    const int node0c = nchild - 1;
    fgate_kernel<<<dim3(2 * nchild, 2), 256, 0, stream>>>(hbuf, cbuf, Ufb, U_f_b,
                                                          fc, nchild, node0c);
    iou_kernel<<<dim3(2 * nnodes, 4), 256, 0, stream>>>(hbuf, fc, Ub, b_iou,
                                                        hbuf, cbuf,
                                                        nnodes, node0, node0c);
  }

  // 4) node-mean and classifier
  reduce_kernel<<<256, 256, 0, stream>>>(hbuf, rep);
  cls_kernel<<<1, 128, 0, stream>>>(rep, cls_w, cls_b, out);

  (void)in_sizes; (void)n_in; (void)out_size; (void)ws_size;
}
